// RNNModel_44435731644620
// MI455X (gfx1250) — matrix operand order repaired and checked by reading, not compiled, not run on hardware
//
#include <hip/hip_runtime.h>
#include <hip/hip_bf16.h>

// ---------------------------------------------------------------------------
// Bidirectional 2-layer LSTM for MI455X (gfx1250), wave32 + WMMA bf16.
// B=32, T=512, D_IN=1024, H=512, 4H=2048, L=2, NDIR=2.
// ---------------------------------------------------------------------------

#define BATCH 32
#define SEQ   512
#define DIN   1024
#define HID   512
#define GATES 2048          // 4*HID
#define ROWS  (SEQ * BATCH) // 16384 GEMM rows (time-major)

typedef unsigned short u16t;
typedef __attribute__((ext_vector_type(16))) __bf16 v16bf;
typedef __attribute__((ext_vector_type(8)))  float  v8f;

static __device__ __forceinline__ u16t f2bf(float f) {
    unsigned u = __float_as_uint(f);
    unsigned r = u + 0x7FFFu + ((u >> 16) & 1u);   // round-to-nearest-even
    return (u16t)(r >> 16);
}

static __device__ __forceinline__ v8f z8() {
    v8f z = {0.f, 0.f, 0.f, 0.f, 0.f, 0.f, 0.f, 0.f};
    return z;
}

// 16-bit A-matrix 16x32 tile (ISA 7.12.2): lanes 0-15 hold K 0..7 / 16..23,
// lanes 16-31 hold K 8..15 / 24..31 -> two contiguous 16B chunks per lane,
// second at +16 halves.  p already includes row & (lane>>4)*8 k-offset.
static __device__ __forceinline__ v16bf ldA(const u16t* p) {
    union { uint4 q[2]; v16bf v; } u;
    u.q[0] = *(const uint4*)(p);
    u.q[1] = *(const uint4*)(p + 16);
    return u.v;
}

// Pre-packed B tile: each lane's 16 halves are contiguous (32B).
static __device__ __forceinline__ v16bf ldB(const u16t* p) {
    union { uint4 q[2]; v16bf v; } u;
    u.q[0] = *(const uint4*)(p);
    u.q[1] = *(const uint4*)(p + 8);
    return u.v;
}

#define WMMA_BF16(a, b, c) \
    __builtin_amdgcn_wmma_f32_16x16x32_bf16(false, (a), false, (b), (short)0, (c), false, false)

static __device__ __forceinline__ float sigm(float x) {
    return 1.0f / (1.0f + __expf(-x));
}

// ---------------------------------------------------------------------------
// bias sum: bsum[l][d][g] = bx + bh   (4*2048 elements)
// ---------------------------------------------------------------------------
__global__ void k_bias(const float* __restrict__ bx, const float* __restrict__ bh,
                       float* __restrict__ bsum, int n) {
    int i = blockIdx.x * blockDim.x + threadIdx.x;
    if (i < n) bsum[i] = bx[i] + bh[i];
}

// ---------------------------------------------------------------------------
// x [B,T,D] fp32 -> xbf [T*B, D] bf16 (time-major rows t*B + b)
// ---------------------------------------------------------------------------
__global__ void k_cvt_x(const float* __restrict__ x, u16t* __restrict__ xbf) {
    int o = blockIdx.x * blockDim.x + threadIdx.x;   // output index
    if (o >= ROWS * DIN) return;
    int d   = o & (DIN - 1);
    int row = o >> 10;          // DIN = 1024
    int b   = row & (BATCH - 1);
    int t   = row >> 5;
    xbf[o] = f2bf(x[((size_t)b * SEQ + t) * DIN + d]);
}

// ---------------------------------------------------------------------------
// Pack a fp32 weight [K, 2048] into per-lane WMMA B layout (bf16):
//   blob(nt,kt) = 32 lanes x 16 halves; lane holds N = nt*16 + (lane&15),
//   K = kt*32 + (lane>>4)*8 + (h&7) + (h>>3)*16  for h = 0..15.
// ---------------------------------------------------------------------------
__global__ void k_pack(const float* __restrict__ W, u16t* __restrict__ P,
                       int kblk, int total) {
    int idx = blockIdx.x * blockDim.x + threadIdx.x;
    if (idx >= total) return;
    int blob   = idx >> 9;        // 512 halves per blob
    int within = idx & 511;
    int lane   = within >> 4;
    int h      = within & 15;
    int nt = blob / kblk;
    int kt = blob - nt * kblk;
    int n = nt * 16 + (lane & 15);
    int k = kt * 32 + ((lane >> 4) << 3) + (h & 7) + ((h >> 3) << 4);
    P[idx] = f2bf(W[(size_t)k * GATES + n]);
}

// ---------------------------------------------------------------------------
// Input projection: gx[dir][row][g] = A[row][:] @ Wx_bf16 + (bx+bh)
//   A: [16384, 1024] bf16 row-major.  grid = (N/64, M/128, NDIR), 256 thr.
//   Each wave: one 16-row M tile x four 16-col N tiles, K = 1024.
// ---------------------------------------------------------------------------
__global__ void __launch_bounds__(256)
k_proj(const u16t* __restrict__ A, const u16t* __restrict__ BpBase,
       const float* __restrict__ bsumBase, float* __restrict__ gxBase) {
    const int dir = blockIdx.z;
    const u16t*  Bp   = BpBase   + (size_t)dir * DIN * GATES;
    const float* bsum = bsumBase + dir * GATES;
    float*       gx   = gxBase   + (size_t)dir * ROWS * GATES;

    const int wv   = threadIdx.x >> 5;
    const int lane = threadIdx.x & 31;
    const int m0   = blockIdx.y * 128 + wv * 16;
    const int n0   = blockIdx.x * 64;
    const int ntb  = blockIdx.x * 4;

    v8f acc0 = z8(), acc1 = z8(), acc2 = z8(), acc3 = z8();

    const int arow = m0 + (lane & 15);
    const int kofs = (lane >> 4) << 3;

    for (int kt = 0; kt < DIN / 32; ++kt) {
        v16bf va = ldA(A + (size_t)arow * DIN + kt * 32 + kofs);
        const u16t* b0 = Bp + ((size_t)((ntb + 0) * (DIN / 32) + kt) * 32 + lane) * 16;
        const u16t* b1 = Bp + ((size_t)((ntb + 1) * (DIN / 32) + kt) * 32 + lane) * 16;
        const u16t* b2 = Bp + ((size_t)((ntb + 2) * (DIN / 32) + kt) * 32 + lane) * 16;
        const u16t* b3 = Bp + ((size_t)((ntb + 3) * (DIN / 32) + kt) * 32 + lane) * 16;
        if (kt + 1 < DIN / 32)
            __builtin_prefetch(Bp + ((size_t)(ntb * (DIN / 32) + kt + 1) * 32 + lane) * 16, 0, 0);
        acc0 = WMMA_BF16(va, ldB(b0), acc0);
        acc1 = WMMA_BF16(va, ldB(b1), acc1);
        acc2 = WMMA_BF16(va, ldB(b2), acc2);
        acc3 = WMMA_BF16(va, ldB(b3), acc3);
    }

    // D layout: lane = n | ((m>>3)<<4), vgpr r -> m = r + 8*(lane>>4)
    const int mh = (lane >> 4) * 8;
    const int nc = lane & 15;
    v8f accs[4] = {acc0, acc1, acc2, acc3};
    for (int nti = 0; nti < 4; ++nti) {
        int g = n0 + nti * 16 + nc;
        float bias = bsum[g];
        for (int r = 0; r < 8; ++r) {
            int m = m0 + mh + r;
            gx[(size_t)m * GATES + g] = accs[nti][r] + bias;
        }
    }
}

// ---------------------------------------------------------------------------
// Recurrent scan: one workgroup per direction (1024 thr = 32 waves).
// LDS: h (bf16, padded stride 520) + gates (fp32, stride 2052)  ~ 289 KB.
// c state lives in registers (16 floats / thread).
// ---------------------------------------------------------------------------
#define HB_STR 520
#define G_STR  2052
#define SMEM_BYTES (BATCH * HB_STR * 2 + BATCH * G_STR * 4)

__global__ void __launch_bounds__(1024)
k_scan(const float* __restrict__ gxBase, const u16t* __restrict__ WhPBase,
       u16t* __restrict__ outBf, float* __restrict__ outF,
       float* __restrict__ hn, float* __restrict__ cn, int layer) {
    extern __shared__ __align__(16) char smem[];
    u16t*  hbf = (u16t*)smem;
    float* gsm = (float*)(smem + BATCH * HB_STR * 2);

    const int dir = blockIdx.x;
    const float* gx = gxBase  + (size_t)dir * ROWS * GATES;
    const u16t*  Wp = WhPBase + (size_t)dir * HID * GATES;

    const int tid  = threadIdx.x;
    const int wv   = tid >> 5;
    const int lane = tid & 31;
    const int n0   = wv * 64;
    const int kofs = (lane >> 4) << 3;
    const int mh   = (lane >> 4) * 8;
    const int nc   = lane & 15;

    // zero h state in LDS
    for (int i = tid; i < BATCH * HB_STR; i += 1024) hbf[i] = 0;
    float c[16];
    for (int e = 0; e < 16; ++e) c[e] = 0.0f;
    __syncthreads();

    for (int s = 0; s < SEQ; ++s) {
        const int t = dir ? (SEQ - 1 - s) : s;

        // ---- gates += h @ Wh : 2 M-tiles x 4 N-tiles x 16 K-tiles of WMMA
        v8f a0 = z8(), a1 = z8(), a2 = z8(), a3 = z8();
        v8f b0 = z8(), b1 = z8(), b2 = z8(), b3 = z8();
        const u16t* pa = hbf + (lane & 15) * HB_STR + kofs;
        for (int kt = 0; kt < HID / 32; ++kt) {
            v16bf va0 = ldA(pa + kt * 32);
            v16bf va1 = ldA(pa + 16 * HB_STR + kt * 32);
            v16bf vb0 = ldB(Wp + ((size_t)((wv * 4 + 0) * (HID / 32) + kt) * 32 + lane) * 16);
            v16bf vb1 = ldB(Wp + ((size_t)((wv * 4 + 1) * (HID / 32) + kt) * 32 + lane) * 16);
            v16bf vb2 = ldB(Wp + ((size_t)((wv * 4 + 2) * (HID / 32) + kt) * 32 + lane) * 16);
            v16bf vb3 = ldB(Wp + ((size_t)((wv * 4 + 3) * (HID / 32) + kt) * 32 + lane) * 16);
            a0 = WMMA_BF16(va0, vb0, a0);  b0 = WMMA_BF16(va1, vb0, b0);
            a1 = WMMA_BF16(va0, vb1, a1);  b1 = WMMA_BF16(va1, vb1, b1);
            a2 = WMMA_BF16(va0, vb2, a2);  b2 = WMMA_BF16(va1, vb2, b2);
            a3 = WMMA_BF16(va0, vb3, a3);  b3 = WMMA_BF16(va1, vb3, b3);
        }

        // ---- add precomputed gx (bias included), stage full 32x2048 in LDS
        const float* gxrow = gx + (size_t)t * BATCH * GATES;
        v8f lo[4] = {a0, a1, a2, a3};
        v8f hi[4] = {b0, b1, b2, b3};
        for (int nti = 0; nti < 4; ++nti) {
            int g = n0 + nti * 16 + nc;
            for (int r = 0; r < 8; ++r) {
                int m = mh + r;
                gsm[m * G_STR + g]        = lo[nti][r] + gxrow[(size_t)m * GATES + g];
                gsm[(m + 16) * G_STR + g] = hi[nti][r] + gxrow[(size_t)(m + 16) * GATES + g];
            }
        }
        __syncthreads();

        // ---- fused LSTM cell elementwise (torch gate order i,f,g,o)
        for (int e = 0; e < 16; ++e) {
            int idx = e * 1024 + tid;
            int b = idx >> 9;          // HID = 512
            int j = idx & (HID - 1);
            const float* gr = gsm + b * G_STR;
            float gi = sigm(gr[j]);
            float gf = sigm(gr[HID + j]);
            float gg = tanhf(gr[2 * HID + j]);
            float go = sigm(gr[3 * HID + j]);
            float cc = c[e] * gf + gi * gg;
            c[e] = cc;
            float hh = go * tanhf(cc);
            hbf[b * HB_STR + j] = f2bf(hh);
            if (outBf) {   // layer 0: bf16 time-major [T*B, 2H] for next layer's GEMM
                outBf[(size_t)(t * BATCH + b) * (2 * HID) + dir * HID + j] = f2bf(hh);
            } else {       // layer 1: fp32 final output [B, T, 2H]
                outF[((size_t)b * SEQ + t) * (2 * HID) + dir * HID + j] = hh;
            }
            if (s == SEQ - 1) {
                size_t o = ((size_t)(layer * 2 + dir) * BATCH + b) * HID + j;
                hn[o] = hh;
                cn[o] = cc;
            }
        }
        __syncthreads();
    }
}

// ---------------------------------------------------------------------------
extern "C" void kernel_launch(void* const* d_in, const int* in_sizes, int n_in,
                              void* d_out, int out_size, void* d_ws, size_t ws_size,
                              hipStream_t stream) {
    (void)in_sizes; (void)n_in; (void)out_size; (void)ws_size;
    const float* x  = (const float*)d_in[0];  // [B,T,DIN]
    const float* Wx = (const float*)d_in[1];  // [L,ND,DIN,4H]
    const float* bx = (const float*)d_in[2];  // [L,ND,4H]
    const float* Wh = (const float*)d_in[3];  // [L,ND,H,4H]
    const float* bh = (const float*)d_in[4];  // [L,ND,4H]
    float* out = (float*)d_out;

    char* ws = (char*)d_ws;
    const size_t SZ_XBF  = (size_t)ROWS * DIN * 2;          // 32 MB
    const size_t SZ_HS0  = (size_t)ROWS * (2 * HID) * 2;    // 32 MB
    const size_t SZ_WXP1 = (size_t)DIN * GATES * 2;         // per (l,d), 4 MB
    const size_t SZ_WHP1 = (size_t)HID * GATES * 2;         // per (l,d), 2 MB
    u16t*  xbf   = (u16t*)(ws);
    u16t*  hs0   = (u16t*)(ws + SZ_XBF);
    u16t*  wxp   = (u16t*)(ws + SZ_XBF + SZ_HS0);
    u16t*  whp   = (u16t*)(ws + SZ_XBF + SZ_HS0 + 4 * SZ_WXP1);
    float* bsum  = (float*)(ws + SZ_XBF + SZ_HS0 + 4 * SZ_WXP1 + 4 * SZ_WHP1);
    float* gx    = (float*)(ws + SZ_XBF + SZ_HS0 + 4 * SZ_WXP1 + 4 * SZ_WHP1 + 4 * GATES * 4);

    float* hn = out + (size_t)BATCH * SEQ * 2 * HID;        // [4,B,H]
    float* cn = hn + (size_t)4 * BATCH * HID;               // [4,B,H]

    // 1) bias sums
    k_bias<<<(4 * GATES + 255) / 256, 256, 0, stream>>>(bx, bh, bsum, 4 * GATES);
    // 2) x -> bf16 time-major
    k_cvt_x<<<(ROWS * DIN) / 256, 256, 0, stream>>>(x, xbf);
    // 3) pack weights into WMMA B layout (4x Wx, 4x Wh)
    for (int ld = 0; ld < 4; ++ld) {
        k_pack<<<(DIN * GATES + 255) / 256, 256, 0, stream>>>(
            Wx + (size_t)ld * DIN * GATES, wxp + (size_t)ld * DIN * GATES,
            DIN / 32, DIN * GATES);
        k_pack<<<(HID * GATES + 255) / 256, 256, 0, stream>>>(
            Wh + (size_t)ld * HID * GATES, whp + (size_t)ld * HID * GATES,
            HID / 32, HID * GATES);
    }

    dim3 pgrid(GATES / 64, ROWS / 128, 2);
    // 4) layer 0: projection + scan (writes bf16 hidden buffer)
    k_proj<<<pgrid, 256, 0, stream>>>(xbf, wxp, bsum, gx);
    k_scan<<<2, 1024, SMEM_BYTES, stream>>>(gx, whp, hs0, nullptr, hn, cn, 0);
    // 5) layer 1: projection (input = layer-0 hidden) + scan (writes d_out)
    k_proj<<<pgrid, 256, 0, stream>>>(hs0, wxp + 2 * (size_t)DIN * GATES,
                                      bsum + 2 * GATES, gx);
    k_scan<<<2, 1024, SMEM_BYTES, stream>>>(gx, whp + 2 * (size_t)HID * GATES,
                                            nullptr, out, hn, cn, 1);
}
